// Qwen3VLTextAttention_44564580663467
// MI455X (gfx1250) — matrix operand order repaired and checked by reading, not compiled, not run on hardware
//
#include <hip/hip_runtime.h>
#include <hip/hip_bf16.h>
#include <stdint.h>

// ---------------------------------------------------------------------------
// Qwen3 attention block for gfx1250 (MI455X), wave32 + WMMA f16->f32.
// GEMMs: LDS double-buffered via async global->LDS copies (ASYNCcnt).
// ---------------------------------------------------------------------------

typedef __attribute__((ext_vector_type(16))) _Float16 v16h;
typedef __attribute__((ext_vector_type(8)))  float    v8f;
typedef __attribute__((ext_vector_type(4)))  int      v4i;

union F16x16 { v16h h; uint32_t u[8]; };

constexpr int S_  = 2048;
constexpr int H_  = 2048;
constexpr int NH_ = 16;
constexpr int NKV_= 8;
constexpr int HD_ = 128;

static __device__ __forceinline__ v8f wmma_f16(const v16h& a, const v16h& b, const v8f& c) {
    // D = A(16x32 f16) * B(32x16 f16) + C(16x16 f32)
    return __builtin_amdgcn_wmma_f32_16x16x32_f16(false, a, false, b, (short)0, c, false, false);
}

// ---- async global->LDS copy (gfx1250) with compile-safe fallback ----------
#if defined(__has_builtin)
#if __has_builtin(__builtin_amdgcn_global_load_async_to_lds_b128)
#define HAVE_ASYNC_LDS 1
#endif
#endif

#ifdef HAVE_ASYNC_LDS
typedef __attribute__((address_space(1))) v4i gv4i_t;   // global int4*
typedef __attribute__((address_space(3))) v4i sv4i_t;   // LDS int4*
static __device__ __forceinline__ void cp16(const _Float16* g, _Float16* l) {
    __builtin_amdgcn_global_load_async_to_lds_b128((gv4i_t*)g, (sv4i_t*)l, 0, 0);
}
static __device__ __forceinline__ void wait_async_6() {
#if __has_builtin(__builtin_amdgcn_s_wait_asynccnt)
    __builtin_amdgcn_s_wait_asynccnt(6);
#else
    __asm__ volatile("s_wait_asynccnt 0x6" ::: "memory");
#endif
}
static __device__ __forceinline__ void wait_async_0() {
#if __has_builtin(__builtin_amdgcn_s_wait_asynccnt)
    __builtin_amdgcn_s_wait_asynccnt(0);
#else
    __asm__ volatile("s_wait_asynccnt 0x0" ::: "memory");
#endif
}
#else
static __device__ __forceinline__ void cp16(const _Float16* g, _Float16* l) {
    *(uint4*)l = *(const uint4*)g;
}
static __device__ __forceinline__ void wait_async_6() {}
static __device__ __forceinline__ void wait_async_0() {}
#endif

// ---------------------------------------------------------------------------
// Elementwise cast f32 -> f16
// ---------------------------------------------------------------------------
__global__ void cast_f32_to_f16(const float* __restrict__ in, _Float16* __restrict__ out, int n) {
    for (int i = blockIdx.x * blockDim.x + threadIdx.x; i < n; i += gridDim.x * blockDim.x)
        out[i] = (_Float16)in[i];
}

// W[K][N] (row-major f32) -> WT[N][K] (row-major f16)
__global__ void transpose_cast_f16(const float* __restrict__ W, _Float16* __restrict__ WT,
                                   int K, int N) {
    int total = K * N;
    for (int i = blockIdx.x * blockDim.x + threadIdx.x; i < total; i += gridDim.x * blockDim.x) {
        int n = i / K, k = i - n * K;
        WT[i] = (_Float16)W[(size_t)k * N + n];
    }
}

// V f32 [S, NKV*HD] -> Vt f16 [NKV][HD][S]
__global__ void v_cast_transpose(const float* __restrict__ Vf, _Float16* __restrict__ Vt) {
    int total = NKV_ * HD_ * S_;
    for (int i = blockIdx.x * blockDim.x + threadIdx.x; i < total; i += gridDim.x * blockDim.x) {
        int h = i / (HD_ * S_);
        int r = i - h * (HD_ * S_);
        int d = r / S_;
        int s = r - d * S_;
        Vt[i] = (_Float16)Vf[((size_t)s * NKV_ + h) * HD_ + d];
    }
}

// ---------------------------------------------------------------------------
// WMMA GEMM with LDS double buffering:
//   C[M,N] f32 = A[M,K] f16 * BT[N,K]^T f16
// Block: 256 threads = 8 waves stacked along M. Block tile 128(M) x 64(N).
// K staged in chunks of 64 halves; async copies fill buf^1 while buf computes.
// LDS chunks (16B) are XOR-swizzled: chunk col ^= (row & 7) -> conflict-free
// fragment reads.
// Grid: (N/64, M/128).
// ---------------------------------------------------------------------------
__global__ __launch_bounds__(256) void wmma_gemm_f16(
    const _Float16* __restrict__ A, const _Float16* __restrict__ BT,
    float* __restrict__ C, int M, int N, int K)
{
    __shared__ __align__(16) _Float16 sA[2][128 * 64];   // 2 x 16 KB
    __shared__ __align__(16) _Float16 sB[2][64 * 64];    // 2 x  8 KB

    const int tid  = threadIdx.x;
    const int lane = tid & 31;
    const int wave = tid >> 5;
    const int l16  = lane & 15;
    const int half = lane >> 4;

    const int mBase = blockIdx.y * 128;
    const int n0    = blockIdx.x * 64;
    const int m0    = mBase + wave * 16;

    v8f acc[4] = {};

    auto stage = [&](int buf, int kk) {
        // A tile: 128 rows x 64 halves = 1024 x 16B chunks, 4 per thread
#pragma unroll
        for (int t = 0; t < 4; ++t) {
            int c   = tid + t * 256;
            int row = c >> 3, col = c & 7;
            const _Float16* g = A + (size_t)(mBase + row) * K + kk + col * 8;
            _Float16*       l = sA[buf] + (row * 8 + (col ^ (row & 7))) * 8;
            cp16(g, l);
        }
        // B tile: 64 rows x 64 halves = 512 x 16B chunks, 2 per thread
#pragma unroll
        for (int t = 0; t < 2; ++t) {
            int c   = tid + t * 256;
            int row = c >> 3, col = c & 7;
            const _Float16* g = BT + (size_t)(n0 + row) * K + kk + col * 8;
            _Float16*       l = sB[buf] + (row * 8 + (col ^ (row & 7))) * 8;
            cp16(g, l);
        }
    };

    stage(0, 0);

    const int rowA  = wave * 16 + l16;     // A-frag row for this lane
    const int maskA = rowA & 7;

    for (int kk = 0; kk < K; kk += 64) {
        const int cur = (kk >> 6) & 1;
        const bool hasNext = (kk + 64) < K;
        if (hasNext) stage(cur ^ 1, kk + 64);
        if (hasNext) wait_async_6(); else wait_async_0();
        __syncthreads();

        const uint32_t* pA = (const uint32_t*)sA[cur];
        const uint32_t* pB = (const uint32_t*)sB[cur];

#pragma unroll
        for (int ks = 0; ks < 64; ks += 32) {
            // A fragment: 16x32 f16, ISA layout (lanes 0-15: K base 0, 16-31: K base 8)
            F16x16 af;
            {
                const int c0 = ((ks >> 3) + half) ^ maskA;        // swizzled 16B chunk
                const int c1 = ((ks >> 3) + half + 2) ^ maskA;
                const int rb = rowA * 32;
#pragma unroll
                for (int i = 0; i < 4; ++i) af.u[i]     = pA[rb + c0 * 4 + i];
#pragma unroll
                for (int i = 0; i < 4; ++i) af.u[4 + i] = pA[rb + c1 * 4 + i];
            }
#pragma unroll
            for (int t = 0; t < 4; ++t) {
                // B fragment: 32x16 f16, ISA layout (lanes 0-15: K base 0, 16-31: K base 8)
                const int rowB  = 16 * t + l16;
                const int maskB = rowB & 7;
                const int c0 = ((ks >> 3) + half) ^ maskB;
                const int c1 = ((ks >> 3) + half + 2) ^ maskB;
                const int rb = rowB * 32;
                F16x16 bf;
#pragma unroll
                for (int i = 0; i < 4; ++i) bf.u[i]     = pB[rb + c0 * 4 + i];
#pragma unroll
                for (int i = 0; i < 4; ++i) bf.u[4 + i] = pB[rb + c1 * 4 + i];
                acc[t] = wmma_f16(af.h, bf.h, acc[t]);
            }
        }
        __syncthreads();
    }

#pragma unroll
    for (int t = 0; t < 4; ++t)
#pragma unroll
        for (int r = 0; r < 8; ++r)
            C[(size_t)(m0 + r + 8 * half) * N + (n0 + 16 * t + l16)] = acc[t][r];
}

// ---------------------------------------------------------------------------
// Per-(token, head) RMS norm + RoPE + cast to f16.
// grid.x = S * (NH + NKV); block = 128 (= HD)
// ---------------------------------------------------------------------------
__global__ __launch_bounds__(128) void norm_rope_cast(
    const float* __restrict__ Qf, const float* __restrict__ Kf,
    const float* __restrict__ cosb, const float* __restrict__ sinb,
    const float* __restrict__ qw, const float* __restrict__ kw,
    _Float16* __restrict__ Qh, _Float16* __restrict__ Kh)
{
    __shared__ float red[HD_];
    __shared__ float xs[HD_];

    int bid = blockIdx.x;
    int s   = bid / (NH_ + NKV_);
    int hh  = bid - s * (NH_ + NKV_);
    int d   = threadIdx.x;
    bool isq = (hh < NH_);

    float x = isq ? Qf[((size_t)s * NH_  + hh)        * HD_ + d]
                  : Kf[((size_t)s * NKV_ + (hh - NH_)) * HD_ + d];

    red[d] = x * x;
    __syncthreads();
    for (int off = 64; off > 0; off >>= 1) {
        if (d < off) red[d] += red[d + off];
        __syncthreads();
    }
    float inv = rsqrtf(red[0] * (1.0f / HD_) + 1e-6f);
    float w   = isq ? qw[d] : kw[d];
    float nrm = x * inv * w;

    xs[d] = nrm;
    __syncthreads();
    float rot = (d < 64) ? -xs[d + 64] : xs[d - 64];

    float c  = cosb[(size_t)s * HD_ + d];
    float si = sinb[(size_t)s * HD_ + d];
    _Float16 out = (_Float16)(nrm * c + rot * si);

    if (isq) Qh[((size_t)s * NH_ + hh) * HD_ + d] = out;
    else     Kh[((size_t)(hh - NH_) * S_ + s) * HD_ + d] = out;
}

// ---------------------------------------------------------------------------
// Flash attention: one wave per (head, 16-query tile).
// Qh [S, NH, HD] f16 ; Kh [NKV, S, HD] f16 ; Vt [NKV, HD, S] f16
// O  [S, NH*HD] f16
// ---------------------------------------------------------------------------
__global__ __launch_bounds__(256) void flash_attn(
    const _Float16* __restrict__ Qh, const _Float16* __restrict__ Kh,
    const _Float16* __restrict__ Vt, const unsigned char* __restrict__ kvmask,
    _Float16* __restrict__ O)
{
    __shared__ __align__(16) _Float16 sP[8][16 * 32];   // per-wave probs tile (16 rows x 32 cols)

    const int lane = threadIdx.x & 31;
    const int wave = threadIdx.x >> 5;
    const int l16  = lane & 15;
    const int half = lane >> 4;

    const int gw = blockIdx.x * 8 + wave;       // 2048 waves total
    const int h  = gw >> 7;                     // head 0..15
    const int it = gw & 127;                    // i-tile 0..127
    const int i0 = it * 16;
    const int hk = h >> 1;                      // GQA: kv head = h / (NH/NKV)

    const int aBase = half * 8;
    const int bBase = half * 16;
    const float scale = 0.08838834764831845f;   // 1/sqrt(128)

    // Q fragments for full HD=128 (4 k-steps of 32)
    F16x16 qa[4];
    {
        const _Float16* Qp = Qh + ((size_t)(i0 + l16) * NH_ + h) * HD_;
#pragma unroll
        for (int kk = 0; kk < 4; ++kk)
#pragma unroll
            for (int i = 0; i < 8; ++i) {
                int k = kk * 32 + aBase + ((i >> 2) << 4) + ((i & 3) << 1);
                qa[kk].u[i] = *(const uint32_t*)(Qp + k);
            }
    }

    // Row masks (kvmask[i]) for this lane's 8 rows
    float rm[8];
#pragma unroll
    for (int r = 0; r < 8; ++r)
        rm[r] = kvmask[i0 + r + 8 * half] ? 1.0f : 0.0f;

    v8f o[8] = {};
    float m[8], l[8];
#pragma unroll
    for (int r = 0; r < 8; ++r) { m[r] = -1e30f; l[r] = 0.0f; }

    _Float16* sp = sP[wave];
    const uint32_t* spu = (const uint32_t*)sp;

    for (int j0 = 0; j0 < i0 + 16; j0 += 32) {
        // ---- scores: two 16x16 tiles (cols j0..j0+15, j0+16..j0+31) ----
        v8f sc[2] = {};
        float cm[2];
#pragma unroll
        for (int t = 0; t < 2; ++t) {
            int jcol = j0 + 16 * t + l16;
            cm[t] = kvmask[jcol] ? 1.0f : 0.0f;
            const _Float16* Krow = Kh + ((size_t)hk * S_ + jcol) * HD_;
#pragma unroll
            for (int kk = 0; kk < 4; ++kk) {
                F16x16 bf;
#pragma unroll
                for (int i = 0; i < 8; ++i)
                    bf.u[i] = *(const uint32_t*)(Krow + kk * 32 + aBase + ((i >> 2) << 4) + ((i & 3) << 1));
                sc[t] = wmma_f16(qa[kk].h, bf.h, sc[t]);
            }
        }

        // ---- mask + scale ----
        float st[2][8];
#pragma unroll
        for (int t = 0; t < 2; ++t) {
            int jcol = j0 + 16 * t + l16;
#pragma unroll
            for (int r = 0; r < 8; ++r) {
                int irow = i0 + r + 8 * half;
                bool ok = (jcol <= irow) && (cm[t] != 0.0f);
                st[t][r] = ok ? sc[t][r] * scale : -1e9f;
            }
        }

        // ---- online softmax (row reductions across 16-lane halves) ----
        float p0[8], p1[8], alpha[8];
#pragma unroll
        for (int r = 0; r < 8; ++r) {
            float rmax = fmaxf(st[0][r], st[1][r]);
#pragma unroll
            for (int msk = 1; msk < 16; msk <<= 1)
                rmax = fmaxf(rmax, __shfl_xor(rmax, msk, 32));
            float newm = fmaxf(m[r], rmax);
            alpha[r] = __expf(m[r] - newm);
            p0[r] = __expf(st[0][r] - newm);
            p1[r] = __expf(st[1][r] - newm);
            float rs = p0[r] + p1[r];
#pragma unroll
            for (int msk = 1; msk < 16; msk <<= 1)
                rs += __shfl_xor(rs, msk, 32);
            l[r] = l[r] * alpha[r] + rs;
            m[r] = newm;
        }
#pragma unroll
        for (int t = 0; t < 8; ++t)
#pragma unroll
            for (int r = 0; r < 8; ++r)
                o[t][r] *= alpha[r];

        // ---- stage probs through LDS: C-layout -> A-fragment layout ----
#pragma unroll
        for (int r = 0; r < 8; ++r) {
            int row = r + 8 * half;
            sp[row * 32 + l16]      = (_Float16)p0[r];
            sp[row * 32 + 16 + l16] = (_Float16)p1[r];
        }
        __builtin_amdgcn_wave_barrier();
        __asm__ volatile("" ::: "memory");

        F16x16 pa;
#pragma unroll
        for (int i = 0; i < 8; ++i)
            pa.u[i] = spu[l16 * 16 + half * 4 + ((i >> 2) << 3) + (i & 3)];
        __builtin_amdgcn_wave_barrier();

        // ---- O += P * V  (8 d-tiles of 16 cols) ----
#pragma unroll
        for (int t = 0; t < 8; ++t) {
            int d = 16 * t + l16;
            const _Float16* Vrow = Vt + ((size_t)hk * HD_ + d) * S_ + j0;
            F16x16 bf;
#pragma unroll
            for (int i = 0; i < 8; ++i)
                bf.u[i] = *(const uint32_t*)(Vrow + aBase + ((i >> 2) << 4) + ((i & 3) << 1));
            o[t] = wmma_f16(pa.h, bf.h, o[t]);
        }
    }

    // ---- epilogue: normalize, apply query-side kv mask, store f16 ----
#pragma unroll
    for (int r = 0; r < 8; ++r) {
        float inv = rm[r] / l[r];
        int irow = i0 + r + 8 * half;
#pragma unroll
        for (int t = 0; t < 8; ++t)
            O[(size_t)irow * (NH_ * HD_) + h * HD_ + 16 * t + l16] = (_Float16)(o[t][r] * inv);
    }
}

// ---------------------------------------------------------------------------
// Host-side launcher
// ---------------------------------------------------------------------------
extern "C" void kernel_launch(void* const* d_in, const int* in_sizes, int n_in,
                              void* d_out, int out_size, void* d_ws, size_t ws_size,
                              hipStream_t stream) {
    (void)in_sizes; (void)n_in; (void)out_size; (void)ws_size;

    const float* X    = (const float*)d_in[0];
    const float* cosb = (const float*)d_in[1];
    const float* sinb = (const float*)d_in[2];
    const unsigned char* kvmask = (const unsigned char*)d_in[3];
    const float* Wq   = (const float*)d_in[4];
    const float* Wk   = (const float*)d_in[5];
    const float* Wv   = (const float*)d_in[6];
    const float* Wo   = (const float*)d_in[7];
    const float* qw   = (const float*)d_in[8];
    const float* kw   = (const float*)d_in[9];

    char* ws = (char*)d_ws;
    size_t off = 0;
    auto alloc = [&](size_t bytes) { char* p = ws + off; off += bytes; return p; };

    _Float16* Xh   = (_Float16*)alloc((size_t)S_ * H_ * 2);            // 8 MB
    _Float16* WqT  = (_Float16*)alloc((size_t)(NH_ * HD_) * H_ * 2);   // 8 MB
    _Float16* WkT  = (_Float16*)alloc((size_t)(NKV_ * HD_) * H_ * 2);  // 4 MB
    _Float16* WvT  = (_Float16*)alloc((size_t)(NKV_ * HD_) * H_ * 2);  // 4 MB
    _Float16* WoT  = (_Float16*)alloc((size_t)H_ * (NH_ * HD_) * 2);   // 8 MB
    float*    Qf   = (float*)   alloc((size_t)S_ * NH_ * HD_ * 4);     // 16 MB
    float*    Kf   = (float*)   alloc((size_t)S_ * NKV_ * HD_ * 4);    // 8 MB
    float*    Vf   = (float*)   alloc((size_t)S_ * NKV_ * HD_ * 4);    // 8 MB
    _Float16* Qh   = (_Float16*)alloc((size_t)S_ * NH_ * HD_ * 2);     // 8 MB
    _Float16* Kh   = (_Float16*)alloc((size_t)NKV_ * S_ * HD_ * 2);    // 4 MB
    _Float16* Vt   = (_Float16*)alloc((size_t)NKV_ * HD_ * S_ * 2);    // 4 MB
    _Float16* Ah   = (_Float16*)alloc((size_t)S_ * NH_ * HD_ * 2);     // 8 MB

    // 1) precision / layout conversion
    cast_f32_to_f16<<<4096, 256, 0, stream>>>(X, Xh, S_ * H_);
    transpose_cast_f16<<<4096, 256, 0, stream>>>(Wq, WqT, H_, NH_ * HD_);
    transpose_cast_f16<<<4096, 256, 0, stream>>>(Wk, WkT, H_, NKV_ * HD_);
    transpose_cast_f16<<<4096, 256, 0, stream>>>(Wv, WvT, H_, NKV_ * HD_);
    transpose_cast_f16<<<4096, 256, 0, stream>>>(Wo, WoT, NH_ * HD_, H_);

    // 2) QKV projections (WMMA GEMM, LDS double-buffered)
    wmma_gemm_f16<<<dim3((NH_ * HD_) / 64, S_ / 128), 256, 0, stream>>>(Xh, WqT, Qf, S_, NH_ * HD_, H_);
    wmma_gemm_f16<<<dim3((NKV_ * HD_) / 64, S_ / 128), 256, 0, stream>>>(Xh, WkT, Kf, S_, NKV_ * HD_, H_);
    wmma_gemm_f16<<<dim3((NKV_ * HD_) / 64, S_ / 128), 256, 0, stream>>>(Xh, WvT, Vf, S_, NKV_ * HD_, H_);

    // 3) RMS norm + RoPE + cast to f16 ; V -> [NKV, HD, S] f16
    norm_rope_cast<<<S_ * (NH_ + NKV_), 128, 0, stream>>>(Qf, Kf, cosb, sinb, qw, kw, Qh, Kh);
    v_cast_transpose<<<4096, 256, 0, stream>>>(Vf, Vt);

    // 4) flash attention (2048 waves: 16 heads x 128 query tiles, 8 waves/block)
    flash_attn<<<(NH_ * (S_ / 16)) / 8, 256, 0, stream>>>(Qh, Kh, Vt, kvmask, Ah);

    // 5) output projection -> fp32 d_out
    wmma_gemm_f16<<<dim3(H_ / 64, S_ / 128), 256, 0, stream>>>(Ah, WoT, (float*)d_out, S_, H_, NH_ * HD_);
}
